// Frag_atom_cross_attention_13322988552792
// MI455X (gfx1250) — matrix operand order repaired and checked by reading, not compiled, not run on hardware
//
#include <hip/hip_runtime.h>
#include <hip/hip_bf16.h>

#define B_ 64
#define A_ 256
#define L_ 1024
#define D_ 128
#define NEG_SLOPE 0.2f
#define LN_EPS 1e-5f
#define ATTN_STRIDE 1032   // 1024 + 8 bf16 pad -> conflict-free ds_load_b128 A-fragments

typedef __attribute__((ext_vector_type(16))) __bf16 v16bf;
typedef __attribute__((ext_vector_type(8)))  float  v8f;

union Frag16 { v16bf v; uint4 q[2]; unsigned int u[8]; };

__device__ __forceinline__ unsigned int pack_bf16x2(float lo, float hi) {
  unsigned int ulo = __float_as_uint(lo);
  unsigned int uhi = __float_as_uint(hi);
  ulo = (ulo + 0x7FFFu + ((ulo >> 16) & 1u)) >> 16;   // round-to-nearest-even
  uhi = (uhi + 0x7FFFu + ((uhi >> 16) & 1u)) >> 16;
  return (ulo & 0xFFFFu) | (uhi << 16);
}

__device__ __forceinline__ float block_reduce_128(float* red, int j, float v) {
  red[j] = v; __syncthreads();
  for (int s = 64; s > 0; s >>= 1) { if (j < s) red[j] += red[j + s]; __syncthreads(); }
  float r = red[0]; __syncthreads();
  return r;
}

// ---------------------------------------------------------------------------
// Kernel 0: fold align weights through the linear layers; transpose W_nb->bf16
//   u_mol[d] = sum_e W_mol[d][e] * w_align_mol[e]
//   u_nb [d] = sum_e W_nb [d][e] * w_align_atom[e]
//   consts[0] = dot(b_mol, w_align_mol); consts[1] = dot(b_nb, w_align_atom)
//   WnbT[e][k] = bf16(W_nb[k][e])
// ---------------------------------------------------------------------------
__global__ void prep_kernel(const float* __restrict__ W_mol, const float* __restrict__ W_nb,
                            const float* __restrict__ b_mol, const float* __restrict__ b_nb,
                            const float* __restrict__ w_am,  const float* __restrict__ w_aa,
                            float* __restrict__ u_mol, float* __restrict__ u_nb,
                            float* __restrict__ consts, unsigned short* __restrict__ WnbT) {
  __shared__ float red[128];
  int i = blockIdx.x;    // k (row of W)
  int j = threadIdx.x;   // e (col of W)
  float wm = W_mol[i * D_ + j];
  float wn = W_nb[i * D_ + j];
  WnbT[j * D_ + i] = (unsigned short)pack_bf16x2(wn, 0.f);
  float r = block_reduce_128(red, j, wm * w_am[j]);
  if (j == 0) u_mol[i] = r;
  r = block_reduce_128(red, j, wn * w_aa[j]);
  if (j == 0) u_nb[i] = r;
  if (i == 0) {
    r = block_reduce_128(red, j, b_mol[j] * w_am[j]);
    if (j == 0) consts[0] = r;
    r = block_reduce_128(red, j, b_nb[j] * w_aa[j]);
    if (j == 0) consts[1] = r;
  }
}

// ---------------------------------------------------------------------------
// Kernel R: out[row] = dot(X[row,:], u) + c      (one wave per row)
// Optionally also writes a bf16 copy of X (fused down-convert for the GEMM).
// ---------------------------------------------------------------------------
__global__ void rowdot_kernel(const float* __restrict__ X, const float* __restrict__ u,
                              const float* __restrict__ cptr, float* __restrict__ out,
                              unsigned short* __restrict__ xbf, int nrows) {
  int wave = threadIdx.x >> 5, lane = threadIdx.x & 31;
  int row = blockIdx.x * 8 + wave;
  if (row >= nrows) return;
  float4 x  = ((const float4*)(X + (size_t)row * D_))[lane];
  float4 uu = ((const float4*)u)[lane];
  if (xbf) {
    uint2 p;
    p.x = pack_bf16x2(x.x, x.y);
    p.y = pack_bf16x2(x.z, x.w);
    ((uint2*)(xbf + (size_t)row * D_))[lane] = p;
  }
  float s = x.x * uu.x + x.y * uu.y + x.z * uu.z + x.w * uu.w;
  for (int m = 16; m >= 1; m >>= 1) s += __shfl_xor(s, m, 32);
  if (lane == 0) out[row] = s + cptr[0];
}

// ---------------------------------------------------------------------------
// Kernel 1: atm = atom_feature @ W_nb + b_nb, stored TRANSPOSED in bf16:
//   atmT[b][d][l]  (value matrix, B-operand friendly for the ctx GEMM)
// One block = (b, 16 l-rows). 8 waves, wave w owns d-cols [16w, 16w+16).
// A and B fragments are straight b128 bf16 loads (no conversion in the loop).
// ---------------------------------------------------------------------------
__global__ void atm_kernel(const unsigned short* __restrict__ atomf_bf,
                           const unsigned short* __restrict__ WnbT,
                           const float* __restrict__ b_nb,
                           unsigned short* __restrict__ atmT) {
  int bid   = blockIdx.x;
  int b     = bid >> 6;            // L/16 = 64 tiles per batch
  int lbase = (bid & 63) << 4;
  int wave  = threadIdx.x >> 5, lane = threadIdx.x & 31;
  int half  = lane >> 4, l16 = lane & 15;
  int dbase = wave << 4;
  int koffA = half ? 8 : 0;        // ISA A-operand lane layout
  int koffB = half ? 8 : 0;        // ISA B-operand lane layout

  const unsigned short* arow = atomf_bf + ((size_t)(b * L_ + lbase + l16)) * D_ + koffA;
  const unsigned short* bcol = WnbT + (dbase + l16) * D_ + koffB;

  v8f acc = {0.f, 0.f, 0.f, 0.f, 0.f, 0.f, 0.f, 0.f};
  #pragma unroll
  for (int kc = 0; kc < 4; ++kc) {
    int kbase = kc * 32;
    Frag16 a, bm;
    a.q[0] = *(const uint4*)(arow + kbase);          // K koffA..koffA+7
    a.q[1] = *(const uint4*)(arow + kbase + 16);     // K koffA+16..koffA+23
    const uint4* pb = (const uint4*)(bcol + kbase);  // K koffB..koffB+7, koffB+16..koffB+23
    bm.q[0] = pb[0]; bm.q[1] = pb[2];
    acc = __builtin_amdgcn_wmma_f32_16x16x32_bf16(false, a.v, false, bm.v,
                                                  (short)0, acc, false, false);
  }
  float bias = b_nb[dbase + l16];
  int mbase = half ? 8 : 0;        // C/D: lane = col, VGPR = row, halves split M by 8
  uint4 st;
  st.x = pack_bf16x2(acc[0] + bias, acc[1] + bias);
  st.y = pack_bf16x2(acc[2] + bias, acc[3] + bias);
  st.z = pack_bf16x2(acc[4] + bias, acc[5] + bias);
  st.w = pack_bf16x2(acc[6] + bias, acc[7] + bias);
  *(uint4*)(atmT + (size_t)b * (D_ * L_) + (dbase + l16) * L_ + lbase + mbase) = st;
}

// ---------------------------------------------------------------------------
// Kernel 3: fused score + softmax + (attn @ atm) + LayerNorm.
// One block = (b, 16 A-rows). 8 waves; softmax: wave w owns rows 2w,2w+1;
// GEMM: wave w owns d-cols [16w,16w+16), K = 1024 in 32 chunks of 32
// (double-buffered so loads for chunk k+1 overlap the v_wmma of chunk k).
// ---------------------------------------------------------------------------
__global__ void attn_kernel(const unsigned short* __restrict__ atmT,
                            const float* __restrict__ s_mol, const float* __restrict__ s_atm,
                            const float* __restrict__ amask, const float* __restrict__ smask_g,
                            const float* __restrict__ balign_p,
                            const float* __restrict__ gamma, const float* __restrict__ beta,
                            float* __restrict__ out) {
  __shared__ unsigned short attn[16 * ATTN_STRIDE];
  __shared__ float satm[L_];
  __shared__ float smk[L_];
  __shared__ float smol_s[16], am_s[16], rscale_s[16];
  __shared__ float psum[8][16], psq[8][16];
  __shared__ float mu_s[16], rstd_s[16];

  int bid = blockIdx.x;
  int b   = bid >> 4;              // A/16 = 16 tiles per batch
  int a0  = (bid & 15) << 4;
  int tid = threadIdx.x;
  int wave = tid >> 5, lane = tid & 31;
  float balign = balign_p[0];

  // stage the per-l score vectors with b128 traffic (256 thr * float4 = 1024)
  ((float4*)satm)[tid] = ((const float4*)(s_atm   + (size_t)b * L_))[tid];
  ((float4*)smk)[tid]  = ((const float4*)(smask_g + (size_t)b * L_))[tid];
  if (tid < 16) {
    smol_s[tid] = s_mol[b * A_ + a0 + tid] + balign;
    am_s[tid]   = amask[b * A_ + a0 + tid];
  }
  __syncthreads();

  // --- softmax over L per A-row (unnormalized exp stored as bf16 in LDS) ---
  for (int rr = 0; rr < 2; ++rr) {
    int row = wave * 2 + rr;
    float sm = smol_s[row];
    float vmax = -3.0e38f;
    for (int it = 0; it < 32; ++it) {
      int l = it * 32 + lane;
      float x = sm + satm[l];
      x = fmaxf(x, 0.f) + NEG_SLOPE * fminf(x, 0.f);   // leaky relu
      x += smk[l];
      vmax = fmaxf(vmax, x);
    }
    for (int m = 16; m >= 1; m >>= 1) vmax = fmaxf(vmax, __shfl_xor(vmax, m, 32));
    float vsum = 0.f;
    for (int it = 0; it < 32; ++it) {
      int l = it * 32 + lane;
      float x = sm + satm[l];
      x = fmaxf(x, 0.f) + NEG_SLOPE * fminf(x, 0.f);
      x += smk[l];
      float e = __expf(x - vmax);                      // <= 1, bf16-safe
      vsum += e;
      attn[row * ATTN_STRIDE + l] = (unsigned short)pack_bf16x2(e, 0.f);
    }
    for (int m = 16; m >= 1; m >>= 1) vsum += __shfl_xor(vsum, m, 32);
    if (lane == 0) rscale_s[row] = am_s[row] / vsum;   // fold attend-mask + 1/sum
  }
  __syncthreads();

  // --- ctx = attn[16x1024] @ atm[1024 x 16 cols] via WMMA bf16 ---
  int half = lane >> 4, l16 = lane & 15;
  int dbase = wave << 4;
  int koffA = half ? 8 : 0;
  int koffB = half ? 8 : 0;
  const unsigned short* aBase = attn + l16 * ATTN_STRIDE + koffA;
  const unsigned short* bBase = atmT + (size_t)b * (D_ * L_) + (dbase + l16) * L_ + koffB;

  Frag16 aCur, bCur, aNxt, bNxt;
  aCur.q[0] = *(const uint4*)(aBase);
  aCur.q[1] = *(const uint4*)(aBase + 16);
  { const uint4* pb = (const uint4*)(bBase); bCur.q[0] = pb[0]; bCur.q[1] = pb[2]; }

  v8f acc = {0.f, 0.f, 0.f, 0.f, 0.f, 0.f, 0.f, 0.f};
  #pragma unroll
  for (int kc = 0; kc < 32; ++kc) {
    if (kc < 31) {
      int kn = (kc + 1) * 32;
      aNxt.q[0] = *(const uint4*)(aBase + kn);
      aNxt.q[1] = *(const uint4*)(aBase + kn + 16);
      const uint4* pb = (const uint4*)(bBase + kn);
      bNxt.q[0] = pb[0]; bNxt.q[1] = pb[2];
    }
    acc = __builtin_amdgcn_wmma_f32_16x16x32_bf16(false, aCur.v, false, bCur.v,
                                                  (short)0, acc, false, false);
    aCur = aNxt; bCur = bNxt;
  }

  // --- per-row scale, then LayerNorm over D (cross-wave via LDS partials) ---
  int mbase = half ? 8 : 0;
  float vals[8];
  #pragma unroll
  for (int v = 0; v < 8; ++v) vals[v] = acc[v] * rscale_s[mbase + v];

  #pragma unroll
  for (int v = 0; v < 8; ++v) {
    float s = vals[v], q = vals[v] * vals[v];
    for (int m = 8; m >= 1; m >>= 1) {            // reduce within each 16-lane half
      s += __shfl_xor(s, m, 32);
      q += __shfl_xor(q, m, 32);
    }
    if (l16 == 0) { psum[wave][mbase + v] = s; psq[wave][mbase + v] = q; }
  }
  __syncthreads();
  if (tid < 16) {
    float s = 0.f, q = 0.f;
    for (int w = 0; w < 8; ++w) { s += psum[w][tid]; q += psq[w][tid]; }
    float mu  = s * (1.0f / D_);
    float var = q * (1.0f / D_) - mu * mu;
    mu_s[tid]   = mu;
    rstd_s[tid] = rsqrtf(fmaxf(var, 0.f) + LN_EPS);
  }
  __syncthreads();

  int col = dbase + l16;
  float g = gamma[col], bt = beta[col];
  #pragma unroll
  for (int v = 0; v < 8; ++v) {
    int row = mbase + v;
    out[((size_t)(b * A_ + a0 + row)) * D_ + col] =
        (vals[v] - mu_s[row]) * rstd_s[row] * g + bt;
  }
}

// ---------------------------------------------------------------------------
extern "C" void kernel_launch(void* const* d_in, const int* in_sizes, int n_in,
                              void* d_out, int out_size, void* d_ws, size_t ws_size,
                              hipStream_t stream) {
  const float* molf   = (const float*)d_in[0];
  const float* atomf  = (const float*)d_in[1];
  const float* amask  = (const float*)d_in[2];
  const float* smask  = (const float*)d_in[3];
  const float* W_mol  = (const float*)d_in[4];
  const float* b_mol  = (const float*)d_in[5];
  const float* W_nb   = (const float*)d_in[6];
  const float* b_nb   = (const float*)d_in[7];
  const float* w_am   = (const float*)d_in[8];
  const float* w_aa   = (const float*)d_in[9];
  const float* balign = (const float*)d_in[10];
  const float* gamma  = (const float*)d_in[11];
  const float* beta   = (const float*)d_in[12];
  float* out = (float*)d_out;

  char* ws = (char*)d_ws;
  size_t off = 0;
  unsigned short* atmT  = (unsigned short*)(ws + off); off += (size_t)B_ * D_ * L_ * 2;
  unsigned short* atomb = (unsigned short*)(ws + off); off += (size_t)B_ * L_ * D_ * 2;
  unsigned short* WnbT  = (unsigned short*)(ws + off); off += (size_t)D_ * D_ * 2;
  float* u_mol  = (float*)(ws + off); off += D_ * 4;
  float* u_nb   = (float*)(ws + off); off += D_ * 4;
  float* consts = (float*)(ws + off); off += 256;
  float* s_atm  = (float*)(ws + off); off += (size_t)B_ * L_ * 4;
  float* s_mol  = (float*)(ws + off); off += (size_t)B_ * A_ * 4;
  (void)off; (void)ws_size; (void)in_sizes; (void)n_in; (void)out_size;

  prep_kernel<<<D_, D_, 0, stream>>>(W_mol, W_nb, b_mol, b_nb, w_am, w_aa,
                                     u_mol, u_nb, consts, WnbT);
  rowdot_kernel<<<(B_ * L_) / 8, 256, 0, stream>>>(atomf, u_nb, consts + 1, s_atm,
                                                   atomb, B_ * L_);
  rowdot_kernel<<<(B_ * A_) / 8, 256, 0, stream>>>(molf, u_mol, consts + 0, s_mol,
                                                   (unsigned short*)nullptr, B_ * A_);
  atm_kernel<<<B_ * (L_ / 16), 256, 0, stream>>>(atomb, WnbT, b_nb, atmT);
  attn_kernel<<<B_ * (A_ / 16), 256, 0, stream>>>(atmT, s_mol, s_atm, amask, smask,
                                                  balign, gamma, beta, out);
}
